// DeformableTransformer_4020089389516
// MI455X (gfx1250) — matrix operand order repaired and checked by reading, not compiled, not run on hardware
//
#include <hip/hip_runtime.h>

#define B_      16
#define LQ_     300
#define D_      256
#define NH_     8
#define DH_     32
#define NL_     4
#define NPTS_   4
#define DFF_    1024
#define LEN_IN_ 13294
#define KPAD_   320   // attention K padded to multiple of 32
#define VKP_    320   // vhT padded K rows per batch

typedef __attribute__((ext_vector_type(16))) _Float16 v16h;
typedef __attribute__((ext_vector_type(8)))  _Float16 v8h;
typedef __attribute__((ext_vector_type(8)))  float    v8f;
typedef __attribute__((ext_vector_type(4)))  float    v4f;

// ---------------------------------------------------------------------------
// Fragment packing helpers
// ---------------------------------------------------------------------------
__device__ __forceinline__ v16h pack2(v8h a, v8h b) {
  v16h f;
#pragma unroll
  for (int i = 0; i < 8; ++i) { f[i] = a[i]; f[i + 8] = b[i]; }
  return f;
}
__device__ __forceinline__ v16h cvt16(v4f a0, v4f a1, v4f a2, v4f a3) {
  v16h f;
#pragma unroll
  for (int i = 0; i < 4; ++i) {
    f[i]      = (_Float16)a0[i];
    f[i + 4]  = (_Float16)a1[i];
    f[i + 8]  = (_Float16)a2[i];
    f[i + 12] = (_Float16)a3[i];
  }
  return f;
}

// ---- fast loaders: fully in-bounds, 16B-aligned, vectorized ----
__device__ __forceinline__ v16h load_a_fast(const float* p, int kh) {
  const v4f* q0 = (const v4f*)(p + kh);
  const v4f* q1 = (const v4f*)(p + kh + 8);
  return cvt16(q0[0], q0[1], q1[0], q1[1]);
}
__device__ __forceinline__ v16h load_a_fast(const _Float16* p, int kh) {
  return pack2(*(const v8h*)(p + kh), *(const v8h*)(p + kh + 8));
}
__device__ __forceinline__ v16h load_b_fast(const float* p, int kh) {
  const v4f* q = (const v4f*)(p + kh);
  return cvt16(q[0], q[1], q[2], q[3]);
}
__device__ __forceinline__ v16h load_b_fast(const _Float16* p, int kh) {
  const v8h* q = (const v8h*)(p + kh);
  return pack2(q[0], q[1]);
}

// ---- slow loaders: per-element bounds (edge tiles only) ----
template <typename T>
__device__ __forceinline__ v16h load_a_slow(const T* p, long lda, int r, int M,
                                            int k0, int kv) {
  int lane = threadIdx.x & 31;
  int kh = (lane >> 4) * 16;
  bool rok = r < M;
  const T* rp = p + (long)r * lda;
  v16h f;
#pragma unroll
  for (int i = 0; i < 8; ++i) {
    int ka = k0 + kh + i, kb = k0 + kh + 8 + i;
    f[i]     = (rok && ka < kv) ? (_Float16)rp[ka] : (_Float16)0.f;
    f[i + 8] = (rok && kb < kv) ? (_Float16)rp[kb] : (_Float16)0.f;
  }
  return f;
}
template <typename T>
__device__ __forceinline__ v16h load_b_slow(const T* p, long ldbn, int n, int N,
                                            int k0, int kv) {
  int kh = ((threadIdx.x & 31) >> 4) * 16;
  bool nok = n < N;
  const T* np = p + (long)n * ldbn;
  v16h f;
#pragma unroll
  for (int i = 0; i < 16; ++i) {
    int k = k0 + kh + i;
    f[i] = (nok && k < kv) ? (_Float16)np[k] : (_Float16)0.f;
  }
  return f;
}

// ---------------------------------------------------------------------------
// Batched GEMM: C = A(MxK) * W(NxK)^T + bias, K-stride 1 everywhere.
// One wave (single-wave block, full register budget) -> 32x64 strip.
// Branch-free inner loop when the whole strip is in-bounds:
// 2 A frags + 4 B frags -> 8 WMMAs per 32-wide K chunk.
// Generic 16-row body otherwise. TRANSC: store C[n*ldc+m] (contiguous/lane).
// ---------------------------------------------------------------------------
template <typename TA, typename TB, typename TC, bool RELU, bool TRANSC>
__global__ void __launch_bounds__(32, 1)
gemm_wmma_kernel(const TA* __restrict__ A, const TB* __restrict__ Bw,
                 TC* __restrict__ C, const float* __restrict__ bias,
                 int M, int N, int K, int Kv,
                 long lda, long ldbn, long ldc,
                 long bao, long bai, long bbo, long bbi,
                 long bco, long bci, int zinner) {
  int z = blockIdx.z;
  int zo = z / zinner, zi = z - zo * zinner;
  const TA* Ab = A + (long)zo * bao + (long)zi * bai;
  const TB* Bb = Bw + (long)zo * bbo + (long)zi * bbi;
  TC* Cb = C + (long)zo * bco + (long)zi * bci;

  int m0 = blockIdx.x * 32;
  int n0 = blockIdx.y * 64;
  int lane = threadIdx.x & 31;
  int kha = (lane >> 4) * 16;
  int khb = (lane >> 4) * 16;
  int nnl = lane & 15;

  v8f z8 = {0.f, 0.f, 0.f, 0.f, 0.f, 0.f, 0.f, 0.f};

  if ((m0 + 32 <= M) && (n0 + 64 <= N) && (Kv == K)) {
    // ---------------- fully-fast, branch-free path ----------------
    const TA* arow0 = Ab + (long)(m0 + nnl) * lda;
    const TA* arow1 = arow0 + (long)16 * lda;
    const TB* brow0 = Bb + (long)(n0 + nnl) * ldbn;
    v8f acc[8] = {z8, z8, z8, z8, z8, z8, z8, z8};
    for (int k0 = 0; k0 < K; k0 += 32) {
      if (k0 + 32 < K) __builtin_prefetch(arow0 + k0 + 32, 0, 3);
      v16h a0 = load_a_fast(arow0 + k0, kha);
      v16h a1 = load_a_fast(arow1 + k0, kha);
#pragma unroll
      for (int s = 0; s < 4; ++s) {
        v16h b = load_b_fast(brow0 + (long)(s * 16) * ldbn + k0, khb);
        acc[s] = __builtin_amdgcn_wmma_f32_16x16x32_f16(
            false, a0, false, b, (short)0, acc[s], false, false);
        acc[4 + s] = __builtin_amdgcn_wmma_f32_16x16x32_f16(
            false, a1, false, b, (short)0, acc[4 + s], false, false);
      }
    }
    int mb = m0 + 8 * (lane >> 4);
#pragma unroll
    for (int s = 0; s < 4; ++s) {
      int n = n0 + s * 16 + nnl;
      float bi = bias ? bias[n] : 0.f;
#pragma unroll
      for (int half = 0; half < 2; ++half) {
#pragma unroll
        for (int v = 0; v < 8; ++v) {
          int m = mb + half * 16 + v;
          float x = acc[half * 4 + s][v] + bi;
          if (RELU) x = fmaxf(x, 0.f);
          if (TRANSC) Cb[(long)n * ldc + m] = (TC)x;
          else        Cb[(long)m * ldc + n] = (TC)x;
        }
      }
    }
  } else {
    // ---------------- generic edge path: two 16-row halves ----------------
#pragma unroll 1
    for (int half = 0; half < 2; ++half) {
      int mm0 = m0 + half * 16;
      if (mm0 >= M) break;
      int ra = mm0 + nnl;
      bool mfull = (mm0 + 16 <= M);
      int kfull = (Kv >> 5) << 5;
      v8f acc[4] = {z8, z8, z8, z8};

      auto domm = [&](v16h a, int k0, bool slowb) {
#pragma unroll
        for (int s = 0; s < 4; ++s) {
          int nb = n0 + s * 16;
          if (nb < N) {  // wave-uniform
            v16h b;
            if (!slowb && nb + 16 <= N)
              b = load_b_fast(Bb + (long)(nb + nnl) * ldbn + k0, khb);
            else
              b = load_b_slow(Bb, ldbn, nb + nnl, N, k0, Kv);
            acc[s] = __builtin_amdgcn_wmma_f32_16x16x32_f16(
                false, a, false, b, (short)0, acc[s], false, false);
          }
        }
      };

      if (mfull) {
        const TA* arow = Ab + (long)ra * lda;
        for (int k0 = 0; k0 < kfull; k0 += 32)
          domm(load_a_fast(arow + k0, kha), k0, false);
      } else {
        for (int k0 = 0; k0 < kfull; k0 += 32)
          domm(load_a_slow(Ab, lda, ra, M, k0, Kv), k0, false);
      }
      for (int k0 = kfull; k0 < K; k0 += 32)
        domm(load_a_slow(Ab, lda, ra, M, k0, Kv), k0, true);

      int mb = mm0 + 8 * (lane >> 4);
#pragma unroll
      for (int s = 0; s < 4; ++s) {
        int n = n0 + s * 16 + nnl;
        if (n < N) {
          float bi = bias ? bias[n] : 0.f;
#pragma unroll
          for (int v = 0; v < 8; ++v) {
            int m = mb + v;
            if (m < M) {
              float x = acc[s][v] + bi;
              if (RELU) x = fmaxf(x, 0.f);
              if (TRANSC) Cb[(long)n * ldc + m] = (TC)x;
              else        Cb[(long)m * ldc + n] = (TC)x;
            }
          }
        }
      }
    }
  }
}

// ---------------------------------------------------------------------------
// Small utility kernels
// ---------------------------------------------------------------------------
__global__ void add_kernel(const float* __restrict__ a, const float* __restrict__ b,
                           float* __restrict__ o, int n) {
  int i = blockIdx.x * blockDim.x + threadIdx.x;
  if (i < n) o[i] = a[i] + b[i];
}
__global__ void cvt_h_kernel(const float* __restrict__ s, _Float16* __restrict__ d, int n) {
  int i = blockIdx.x * blockDim.x + threadIdx.x;
  if (i < n) d[i] = (_Float16)s[i];
}
__global__ void fill0_kernel(unsigned int* __restrict__ p, int n) {
  int i = blockIdx.x * blockDim.x + threadIdx.x;
  if (i < n) p[i] = 0u;
}

// O = LayerNorm(X + Y) * g + b over last dim 256; one wave per row
__global__ void add_ln_kernel(const float* __restrict__ X, const float* __restrict__ Y,
                              const float* __restrict__ g, const float* __restrict__ be,
                              float* __restrict__ O, int rows) {
  int row = blockIdx.x * (blockDim.x >> 5) + (threadIdx.x >> 5);
  if (row >= rows) return;
  int lane = threadIdx.x & 31;
  const float* xp = X + (long)row * D_;
  const float* yp = Y + (long)row * D_;
  float v[8]; float s = 0.f;
#pragma unroll
  for (int i = 0; i < 8; ++i) { float t = xp[lane + i * 32] + yp[lane + i * 32]; v[i] = t; s += t; }
#pragma unroll
  for (int o = 16; o > 0; o >>= 1) s += __shfl_xor(s, o, 32);
  float mean = s * (1.f / 256.f);
  float var = 0.f;
#pragma unroll
  for (int i = 0; i < 8; ++i) { float d = v[i] - mean; var += d * d; }
#pragma unroll
  for (int o = 16; o > 0; o >>= 1) var += __shfl_xor(var, o, 32);
  float inv = rsqrtf(var * (1.f / 256.f) + 1e-5f);
#pragma unroll
  for (int i = 0; i < 8; ++i) {
    int c = lane + i * 32;
    O[(long)row * D_ + c] = (v[i] - mean) * inv * g[c] + be[c];
  }
}

// softmax over padded rows [rows x KPAD_], valid LQ_, scale 1/sqrt(DH);
// zero-fills K padding (required by P*V GEMM)
__global__ void attn_softmax_kernel(float* __restrict__ p, int rows) {
  int row = blockIdx.x * (blockDim.x >> 5) + (threadIdx.x >> 5);
  if (row >= rows) return;
  int lane = threadIdx.x & 31;
  float* rp = p + (long)row * KPAD_;
  const float scale = 0.17677669529663689f;
  float v[10]; float mx = -3.4e38f;
#pragma unroll
  for (int i = 0; i < 10; ++i) {
    int c = lane + i * 32;
    float t = (c < LQ_) ? rp[c] * scale : -3.4e38f;
    v[i] = t; mx = fmaxf(mx, t);
  }
#pragma unroll
  for (int o = 16; o > 0; o >>= 1) mx = fmaxf(mx, __shfl_xor(mx, o, 32));
  float s = 0.f;
#pragma unroll
  for (int i = 0; i < 10; ++i) {
    int c = lane + i * 32;
    float e = (c < LQ_) ? __expf(v[i] - mx) : 0.f;
    v[i] = e; s += e;
  }
#pragma unroll
  for (int o = 16; o > 0; o >>= 1) s += __shfl_xor(s, o, 32);
  float inv = 1.f / s;
#pragma unroll
  for (int i = 0; i < 10; ++i) rp[lane + i * 32] = v[i] * inv;
}

// MS-deformable sampling: one wave per (b,q,h); lane = channel (DH==32).
__global__ void msdeform_kernel(const float* __restrict__ off,
                                const float* __restrict__ awl,
                                const float* __restrict__ ref,
                                const _Float16* __restrict__ value,
                                float* __restrict__ loc_out,
                                float* __restrict__ aw_out,
                                float* __restrict__ sampled) {
  int wave = blockIdx.x * (blockDim.x >> 5) + (threadIdx.x >> 5);
  int lane = threadIdx.x & 31;
  int h  = wave % NH_;
  int bq = wave / NH_;
  int b  = bq / LQ_;

  const int Hs[4]     = {100, 50, 25, 13};
  const int Ws[4]     = {100, 50, 25, 13};
  const int starts[4] = {0, 10000, 12500, 13125};

  float lg = (lane < 16) ? awl[(long)bq * 128 + h * 16 + lane] : -3.4e38f;
  float mx = lg;
#pragma unroll
  for (int o = 1; o < 16; o <<= 1) mx = fmaxf(mx, __shfl_xor(mx, o, 32));
  float e = (lane < 16) ? __expf(lg - mx) : 0.f;
  float s = e;
#pragma unroll
  for (int o = 1; o < 16; o <<= 1) s += __shfl_xor(s, o, 32);
  float awv = e / s;
  if (lane < 16) aw_out[((long)bq * NH_ + h) * 16 + lane] = awv;

  int j = lane >> 1, c = lane & 1, l = j >> 2;
  float refv = ref[((long)bq * NL_ + l) * 2 + c];
  float offv = off[(long)bq * 256 + h * 32 + lane];
  float nrm  = (c == 0) ? (float)Ws[l] : (float)Hs[l];
  float locv = refv + offv / nrm;
  loc_out[((long)bq * NH_ + h) * 32 + lane] = locv;

  float acc = 0.f;
  for (int pt = 0; pt < 16; ++pt) {
    float x = __shfl(locv, 2 * pt, 32);
    float y = __shfl(locv, 2 * pt + 1, 32);
    float w = __shfl(awv, pt, 32);
    int lv = pt >> 2;
    int Hl = Hs[lv], Wl = Ws[lv];
    float fx = x * (float)Wl - 0.5f;
    float fy = y * (float)Hl - 0.5f;
    float x0f = floorf(fx), y0f = floorf(fy);
    float lx = fx - x0f, ly = fy - y0f;
    int ix = (int)x0f, iy = (int)y0f;
    long base = ((long)b * LEN_IN_ + starts[lv]) * 256 + h * 32 + lane;
    float v00 = 0.f, v01 = 0.f, v10 = 0.f, v11 = 0.f;
    if (iy >= 0 && iy < Hl) {
      long rb = base + (long)iy * Wl * 256;
      if (ix >= 0 && ix < Wl)         v00 = (float)value[rb + (long)ix * 256];
      if (ix + 1 >= 0 && ix + 1 < Wl) v01 = (float)value[rb + (long)(ix + 1) * 256];
    }
    if (iy + 1 >= 0 && iy + 1 < Hl) {
      long rb = base + (long)(iy + 1) * Wl * 256;
      if (ix >= 0 && ix < Wl)         v10 = (float)value[rb + (long)ix * 256];
      if (ix + 1 >= 0 && ix + 1 < Wl) v11 = (float)value[rb + (long)(ix + 1) * 256];
    }
    float bi = v00 * (1.f - lx) * (1.f - ly) + v01 * lx * (1.f - ly)
             + v10 * (1.f - lx) * ly         + v11 * lx * ly;
    acc += w * bi;
  }
  sampled[((long)bq * NH_ + h) * 32 + lane] = acc;
}

// ---------------------------------------------------------------------------
// Host-side launch helpers
// ---------------------------------------------------------------------------
typedef _Float16 f16;
static inline dim3 ggrid(int M, int N, int nz) { return dim3((M + 31) / 32, (N + 63) / 64, nz); }

static inline void gemm_ffh(hipStream_t st, const float* A, const f16* Bw, f16* C,
                            const float* bias, int M, int N, int K,
                            long lda, long ldbn, long ldc) {
  gemm_wmma_kernel<float, f16, f16, false, false><<<ggrid(M, N, 1), 32, 0, st>>>(
      A, Bw, C, bias, M, N, K, K, lda, ldbn, ldc, 0, 0, 0, 0, 0, 0, 1);
}
static inline void gemm_fff(hipStream_t st, const float* A, const f16* Bw, float* C,
                            const float* bias, int M, int N, int K, int Kv,
                            long lda, long ldbn, long ldc,
                            long bao, long bai, long bbo, long bbi, long bco, long bci,
                            int nz, int zinner) {
  gemm_wmma_kernel<float, f16, float, false, false><<<ggrid(M, N, nz), 32, 0, st>>>(
      A, Bw, C, bias, M, N, K, Kv, lda, ldbn, ldc, bao, bai, bbo, bbi, bco, bci, zinner);
}
static inline void gemm_hhf(hipStream_t st, const f16* A, const f16* Bw, float* C,
                            const float* bias, int M, int N, int K,
                            long lda, long ldbn, long ldc,
                            long bao, long bai, long bbo, long bbi, long bco, long bci,
                            int nz, int zinner) {
  gemm_wmma_kernel<f16, f16, float, false, false><<<ggrid(M, N, nz), 32, 0, st>>>(
      A, Bw, C, bias, M, N, K, K, lda, ldbn, ldc, bao, bai, bbo, bbi, bco, bci, zinner);
}
static inline void gemm_ffh_t(hipStream_t st, const float* A, const f16* Bw, f16* C,
                              const float* bias, int M, int N, int K,
                              long lda, long ldbn, long ldc,
                              long bao, long bco, int nz) {
  gemm_wmma_kernel<float, f16, f16, false, true><<<ggrid(M, N, nz), 32, 0, st>>>(
      A, Bw, C, bias, M, N, K, K, lda, ldbn, ldc, bao, 0, 0, 0, bco, 0, 1);
}
static inline void gemm_relu_h(hipStream_t st, const float* A, const f16* Bw, f16* C,
                               const float* bias, int M, int N, int K) {
  gemm_wmma_kernel<float, f16, f16, true, false><<<ggrid(M, N, 1), 32, 0, st>>>(
      A, Bw, C, bias, M, N, K, K, (long)K, (long)K, (long)N, 0, 0, 0, 0, 0, 0, 1);
}

// ---------------------------------------------------------------------------
extern "C" void kernel_launch(void* const* d_in, const int* in_sizes, int n_in,
                              void* d_out, int out_size, void* d_ws, size_t ws_size,
                              hipStream_t stream) {
  (void)in_sizes; (void)n_in; (void)out_size; (void)ws_size;
  const float* tgt      = (const float*)d_in[0];
  const float* qpos     = (const float*)d_in[1];
  const float* ref      = (const float*)d_in[2];
  const float* src      = (const float*)d_in[3];
  const float* sa_in_w  = (const float*)d_in[5];
  const float* sa_in_b  = (const float*)d_in[6];
  const float* sa_out_w = (const float*)d_in[7];
  const float* sa_out_b = (const float*)d_in[8];
  const float* v_w      = (const float*)d_in[9];
  const float* v_b      = (const float*)d_in[10];
  const float* off_w    = (const float*)d_in[11];
  const float* off_b    = (const float*)d_in[12];
  const float* aw_w     = (const float*)d_in[13];
  const float* aw_b     = (const float*)d_in[14];
  const float* o_w      = (const float*)d_in[15];
  const float* o_b      = (const float*)d_in[16];
  const float* l1_w     = (const float*)d_in[17];
  const float* l1_b     = (const float*)d_in[18];
  const float* l2_w     = (const float*)d_in[19];
  const float* l2_b     = (const float*)d_in[20];
  const float* ln1_g    = (const float*)d_in[21];
  const float* ln1_b    = (const float*)d_in[22];
  const float* ln2_g    = (const float*)d_in[23];
  const float* ln2_b    = (const float*)d_in[24];
  const float* ln3_g    = (const float*)d_in[25];
  const float* ln3_b    = (const float*)d_in[26];

  float* out_t   = (float*)d_out;
  float* out_loc = out_t + (long)B_ * LQ_ * D_;
  float* out_aw  = out_loc + (long)B_ * LQ_ * NH_ * NL_ * NPTS_ * 2;

  const int NROW = B_ * LQ_;    // 4800
  const int NELT = NROW * D_;   // 1,228,800

  char* wsp = (char*)d_ws;
  auto alloc = [&](size_t bytes) {
    void* p = (void*)wsp;
    wsp += (bytes + 255) & ~(size_t)255;
    return p;
  };
  // f16 weight slab
  f16* w_sa_in  = (f16*)alloc((size_t)768 * 256 * 2);
  f16* w_sa_out = (f16*)alloc((size_t)256 * 256 * 2);
  f16* w_v      = (f16*)alloc((size_t)256 * 256 * 2);
  f16* w_off    = (f16*)alloc((size_t)256 * 256 * 2);
  f16* w_aw     = (f16*)alloc((size_t)128 * 256 * 2);
  f16* w_o      = (f16*)alloc((size_t)256 * 256 * 2);
  f16* w_l1     = (f16*)alloc((size_t)1024 * 256 * 2);
  f16* w_l2     = (f16*)alloc((size_t)256 * 1024 * 2);
  // activations
  f16*   value  = (f16*)alloc((size_t)B_ * LEN_IN_ * D_ * 2);       // 108.9 MB
  float* qbuf   = (float*)alloc((size_t)NELT * 4);                  // reused as t2q
  f16*   qk     = (f16*)alloc((size_t)NROW * 512 * 2);              // qh|kh (f16)
  f16*   vhT    = (f16*)alloc((size_t)256 * B_ * VKP_ * 2);         // [n][b][k]
  float* probs  = (float*)alloc((size_t)128 * LQ_ * KPAD_ * 4);     // 49.2 MB
  float* attnO  = (float*)alloc((size_t)NELT * 4);                  // reused as sampled
  float* sa_out = (float*)alloc((size_t)NELT * 4);                  // reused as ms_out
  float* t_ln2  = (float*)alloc((size_t)NELT * 4);
  float* offb   = (float*)alloc((size_t)NELT * 4);
  float* awl    = (float*)alloc((size_t)NROW * 128 * 4);
  float* t_ln1  = (float*)alloc((size_t)NELT * 4);
  f16*   ffn1   = (f16*)alloc((size_t)NROW * DFF_ * 2);
  float* ffn2   = (float*)alloc((size_t)NELT * 4);
  float* t2q     = qbuf;
  float* sampled = attnO;
  float* ms_out  = sa_out;

  // ---- 0) weights -> f16 (tiny), zero vhT pad ----
  auto cvtw = [&](const float* s, f16* d, int n) {
    cvt_h_kernel<<<(n + 255) / 256, 256, 0, stream>>>(s, d, n);
  };
  cvtw(sa_in_w, w_sa_in, 768 * 256);
  cvtw(sa_out_w, w_sa_out, 256 * 256);
  cvtw(v_w, w_v, 256 * 256);
  cvtw(off_w, w_off, 256 * 256);
  cvtw(aw_w, w_aw, 128 * 256);
  cvtw(o_w, w_o, 256 * 256);
  cvtw(l1_w, w_l1, 1024 * 256);
  cvtw(l2_w, w_l2, 256 * 1024);
  fill0_kernel<<<(256 * B_ * VKP_ / 2 + 255) / 256, 256, 0, stream>>>(
      (unsigned int*)vhT, 256 * B_ * VKP_ / 2);

  // ---- 1) q = tgt + query_pos ----
  add_kernel<<<NELT / 256, 256, 0, stream>>>(tgt, qpos, qbuf, NELT);

  // ---- 2) [qh|kh] (f16); vhT = (tgt @ Wv.T + b)^T per batch, layout [n][b][320] ----
  gemm_ffh(stream, qbuf, w_sa_in, qk, sa_in_b, NROW, 512, 256, 256, 256, 512);
  gemm_ffh_t(stream, tgt, w_sa_in + 512 * 256, vhT, sa_in_b + 512,
             LQ_, 256, 256, 256, 256, (long)B_ * VKP_,
             (long)LQ_ * 256, (long)VKP_, B_);

  // ---- 3) scores = qh @ kh.T (batched over B*NH), rows padded to 320 ----
  gemm_hhf(stream, qk, qk + 256, probs, nullptr, LQ_, LQ_, 32,
           512, 512, KPAD_,
           (long)LQ_ * 512, 32, (long)LQ_ * 512, 32,
           (long)NH_ * LQ_ * KPAD_, (long)LQ_ * KPAD_, 128, NH_);

  // ---- 4) scaled softmax, zero-fills K pad ----
  attn_softmax_kernel<<<(128 * LQ_) / 4, 128, 0, stream>>>(probs, 128 * LQ_);

  // ---- 5) attnO = P @ V ----
  gemm_fff(stream, probs, vhT, attnO, nullptr, LQ_, DH_, KPAD_, KPAD_,
           KPAD_, (long)B_ * VKP_, 256,
           (long)NH_ * LQ_ * KPAD_, (long)LQ_ * KPAD_,
           (long)VKP_, (long)32 * B_ * VKP_,
           (long)LQ_ * 256, 32, 128, NH_);

  // ---- 6) self-attn out proj; t = LN(tgt + t2) ----
  gemm_fff(stream, attnO, w_sa_out, sa_out, sa_out_b, NROW, 256, 256, 256,
           256, 256, 256, 0, 0, 0, 0, 0, 0, 1, 1);
  add_ln_kernel<<<NROW / 4, 128, 0, stream>>>(tgt, sa_out, ln2_g, ln2_b, t_ln2, NROW);

  // ---- 7) MS-deform inputs ----
  add_kernel<<<NELT / 256, 256, 0, stream>>>(t_ln2, qpos, t2q, NELT);
  gemm_ffh(stream, src, w_v, value, v_b, B_ * LEN_IN_, 256, 256, 256, 256, 256);
  gemm_fff(stream, t2q, w_off, offb, off_b, NROW, 256, 256, 256,
           256, 256, 256, 0, 0, 0, 0, 0, 0, 1, 1);
  gemm_fff(stream, t2q, w_aw, awl, aw_b, NROW, 128, 256, 256,
           256, 256, 128, 0, 0, 0, 0, 0, 0, 1, 1);

  // ---- 8) fused aw-softmax + loc + bilinear sampling ----
  msdeform_kernel<<<(B_ * LQ_ * NH_) / 4, 128, 0, stream>>>(
      offb, awl, ref, value, out_loc, out_aw, sampled);

  // ---- 9) output proj; t = LN(t + t2) ----
  gemm_fff(stream, sampled, w_o, ms_out, o_b, NROW, 256, 256, 256,
           256, 256, 256, 0, 0, 0, 0, 0, 0, 1, 1);
  add_ln_kernel<<<NROW / 4, 128, 0, stream>>>(t_ln2, ms_out, ln1_g, ln1_b, t_ln1, NROW);

  // ---- 10) FFN; t = LN(t + t2) -> final output ----
  gemm_relu_h(stream, t_ln1, w_l1, ffn1, l1_b, NROW, DFF_, 256);
  gemm_hhf(stream, ffn1, w_l2, ffn2, l2_b, NROW, 256, DFF_,
           DFF_, DFF_, 256, 0, 0, 0, 0, 0, 0, 1, 1);
  add_ln_kernel<<<NROW / 4, 128, 0, stream>>>(t_ln1, ffn2, ln3_g, ln3_b, out_t, NROW);
}
